// DeepQNetIVCML_4329327034935
// MI455X (gfx1250) — compile-verified, matrix operand order checked by reading, not run on hardware
//
#include <hip/hip_runtime.h>
#include <hip/hip_bf16.h>

// Problem dims (fixed by the reference)
#define Bb 8
#define Ss 8
#define Nn 32
#define Vv 8192
#define Dd 768

typedef __attribute__((ext_vector_type(16))) __bf16 v16bf;
typedef __attribute__((ext_vector_type(8)))  float  v8f;
typedef __attribute__((ext_vector_type(4)))  unsigned int u32x4;
typedef __attribute__((ext_vector_type(4)))  int    i32x4;
typedef __attribute__((ext_vector_type(8)))  int    i32x8;

__device__ __forceinline__ __bf16 to_bf16(float x) { return (__bf16)x; }

// Issue a 2D TDM load into LDS. Descriptor group1 (dims/strides) is caller-
// built and loop-invariant; only group0 (lds/global addresses) changes.
__device__ __forceinline__ void tdm_load_b_tile(unsigned ldsAddr,
                                                unsigned long long gAddr,
                                                i32x8 g1) {
  u32x4 g0;
  g0[0] = 1u;                                           // count=1, user mode
  g0[1] = ldsAddr;                                      // lds_addr (bytes)
  g0[2] = (unsigned)gAddr;                              // global_addr[31:0]
  g0[3] = ((unsigned)(gAddr >> 32) & 0x01FFFFFFu) | (2u << 30); // addr | type=2
  const i32x4 gz4 = {0, 0, 0, 0};
#if __has_include(<hip/amd_detail/amd_gfx1250_TDM.h>)
  const i32x8 gz8 = {0, 0, 0, 0, 0, 0, 0, 0};
  __builtin_amdgcn_tensor_load_to_lds(g0, g1, gz4, gz4, gz8, 0);
#else
  __builtin_amdgcn_tensor_load_to_lds(g0, g1, gz4, gz4, 0);
#endif
}

// ---------------------------------------------------------------------------
// Tiled GEMM: C[z] = act(A[z] @ Bm[z] + bias)
//   - fp32 row-major A [M x K] (lda), Bm [K x N] (ldb), C (ldc)
//   - grid = (N/128, M/32, batch); block = 256 threads = 8 wave32
//   - wave w owns the 32x16 C tile at cols [nbase+16w, +16): 2 accumulators
//   - K multiple of 32 (8192 / 768 / 1536 / 2304 all qualify)
// Double-buffered pipeline:
//   A tile (32x32 fp32) cooperative float4 load, parked to LDS post-compute.
//   B tile (32x128 fp32) via Tensor Data Mover, issued one tile ahead so the
//   DMA overlaps the WMMA work; s_wait_tensorcnt 0 + one barrier per iter.
// bf16 conversion on LDS read, per CDNA5 16-bit A/B WMMA VGPR striping.
// ---------------------------------------------------------------------------
__global__ __launch_bounds__(256)
void wmma_gemm_kernel(const float* __restrict__ A,
                      const float* __restrict__ Bm,
                      const float* __restrict__ bias,
                      float* __restrict__ C,
                      int K, int lda, int ldb, int ldc,
                      long aStride, long bStride, long cStride,
                      int doRelu)
{
  __shared__ float lA[2 * 32 * 32];    //  8 KB, double buffered
  __shared__ float lB[2 * 32 * 128];   // 32 KB, double buffered

  const long z = blockIdx.z;
  A  += z * aStride;
  Bm += z * bStride;
  C  += z * cStride;

  const int mbase = blockIdx.y * 32;
  const int nbase = blockIdx.x * 128;
  const int tid   = threadIdx.x;
  const int wave  = tid >> 5;
  const int lane  = tid & 31;
  const int half  = lane >> 4;   // 0: lanes 0-15, 1: lanes 16-31
  const int l16   = lane & 15;

  // Loop-invariant addressing (pointer increments only, no per-iter muls)
  const int ar = tid >> 3;            // 0..31
  const int ak = (tid & 7) * 4;       // 0..28
  const float* aPtr = A + (long)(mbase + ar) * lda + ak;
  float* lAdst = &lA[ar * 32 + ak];

  // TDM descriptor group1: 2D tile 128x32 of 4-byte elements, row stride ldb
  const unsigned ldsB = (unsigned)(unsigned long long)(void*)lB;
  unsigned long long gB = (unsigned long long)(const void*)(Bm + nbase);
  const unsigned long long gStep = (unsigned long long)ldb * 32ull * 4ull;

  i32x8 g1;
  g1[0] = (2 << 16);                          // data_size=4B; no mask/pad/iter
  g1[1] = (ldb & 0xFFFF) << 16;               // tensor_dim0[15:0]
  g1[2] = (ldb >> 16) | ((K & 0xFFFF) << 16); // tensor_dim0 hi | dim1 lo
  g1[3] = (K >> 16) | (128 << 16);            // dim1 hi | tile_dim0=128
  g1[4] = 32;                                 // tile_dim1=32, tile_dim2=0
  g1[5] = ldb;                                // tensor_dim0_stride[31:0]
  g1[6] = 0;
  g1[7] = 0;

  const int kIters = K >> 5;

  // ---- prologue: stage tile 0 in buffer 0 --------------------------------
  {
    const float4 a0 = *reinterpret_cast<const float4*>(aPtr);
    aPtr += 32;
    *reinterpret_cast<float4*>(lAdst) = a0;
  }
  if (wave == 0) tdm_load_b_tile(ldsB, gB, g1);
  gB += gStep;

  v8f acc0 = {}, acc1 = {};

  for (int it = 0; it < kIters; ++it) {
    __builtin_amdgcn_s_wait_tensorcnt(0);  // B(it) landed (no-op off wave 0)
    __syncthreads();                       // A(it)/B(it) visible; prev reads done

    const int cur = it & 1;
    const int nxt = cur ^ 1;
    const bool have = (it + 1) < kIters;

    // Kick off tile it+1 (DMA + global load) before computing on tile it.
    float4 aNext = {};
    if (have) {
      aNext = *reinterpret_cast<const float4*>(aPtr);
      aPtr += 32;
      __builtin_prefetch(aPtr, 0, 1);                 // global_prefetch_b8
      if (wave == 0)
        tdm_load_b_tile(ldsB + (unsigned)(nxt * 32 * 128 * 4), gB, g1);
      gB += gStep;
    }

    // ---- bf16 fragments + 2 WMMAs on current buffers ---------------------
    const float* cA = &lA[cur * 32 * 32];
    const float* cB = &lB[cur * 32 * 128];
    v16bf af0, af1, bfv;
#pragma unroll
    for (int j = 0; j < 8; ++j) {
      af0[j]     = to_bf16(cA[l16 * 32 + half * 8 + j]);
      af0[j + 8] = to_bf16(cA[l16 * 32 + 16 + half * 8 + j]);
      af1[j]     = to_bf16(cA[(16 + l16) * 32 + half * 8 + j]);
      af1[j + 8] = to_bf16(cA[(16 + l16) * 32 + 16 + half * 8 + j]);
      bfv[j]     = to_bf16(cB[(half * 8 + j) * 128 + wave * 16 + l16]);
      bfv[j + 8] = to_bf16(cB[(16 + half * 8 + j) * 128 + wave * 16 + l16]);
    }
    acc0 = __builtin_amdgcn_wmma_f32_16x16x32_bf16(
        false, af0, false, bfv, (short)0, acc0, false, false);
    acc1 = __builtin_amdgcn_wmma_f32_16x16x32_bf16(
        false, af1, false, bfv, (short)0, acc1, false, false);

    // Park next A tile after compute (its load has had the math to complete).
    if (have)
      *reinterpret_cast<float4*>(lAdst + nxt * 32 * 32) = aNext;
    // No trailing barrier needed: next iteration's top barrier orders the
    // buffer swap (writes at it+1 target buffer cur, read-complete here).
  }

  // ---- epilogue: bias + relu, scatter per C/D VGPR layout ----------------
#pragma unroll
  for (int r = 0; r < 8; ++r) {
    const int n = nbase + wave * 16 + l16;
    const float bv = bias ? bias[n] : 0.f;
    float v0 = acc0[r] + bv;
    float v1 = acc1[r] + bv;
    if (doRelu) { v0 = fmaxf(v0, 0.f); v1 = fmaxf(v1, 0.f); }
    C[(long)(mbase + half * 8 + r) * ldc + n]      = v0;
    C[(long)(mbase + 16 + half * 8 + r) * ldc + n] = v1;
  }
}

// ---------------------------------------------------------------------------
// q0 = mean(query_fea, axis=-2) into rows 0..7 of qbuf; zero pad rows 8..31
// of X2 (q-update GEMM runs on a 32-row padded A).  grid(32), block(256)
// ---------------------------------------------------------------------------
__global__ __launch_bounds__(256)
void init_kernel(const float* __restrict__ qf, float* __restrict__ q0,
                 float* __restrict__ X2)
{
  const int x = blockIdx.x;
  if (x < Bb) {
    for (int k = threadIdx.x; k < Dd; k += 256) {
      float s = 0.f;
#pragma unroll
      for (int i = 0; i < 16; ++i) s += qf[((long)x * 16 + i) * Dd + k];
      q0[(long)x * Dd + k] = s * (1.f / 16.f);
    }
  } else {
    for (int k = threadIdx.x; k < 3 * Dd; k += 256)
      X2[(long)x * (3 * Dd) + k] = 0.f;
  }
}

// X[r, :] = concat(fn[b, t, n, :], q[b, :]); r = b*32+n.  grid(256), block(256)
__global__ __launch_bounds__(256)
void pack_x_kernel(const float* __restrict__ fn, const float* __restrict__ q,
                   float* __restrict__ X, int t)
{
  const int r = blockIdx.x;
  const int b = r >> 5, n = r & 31;
  const float* src0 = &fn[(((long)b * Ss + t) * Nn + n) * Dd];
  const float* src1 = &q[(long)b * Dd];
  float* dst = &X[(long)r * (2 * Dd)];
  for (int k = threadIdx.x; k < Dd; k += 256) {
    dst[k]      = src0[k];
    dst[Dd + k] = src1[k];
  }
}

// cls[b,t,n] = dot(H[r,:], Wcls) + bcls ; both output copies.
// grid(32), block(256): one wave32 per row.
__global__ __launch_bounds__(256)
void cls_kernel(const float* __restrict__ H, const float* __restrict__ Wcls,
                const float* __restrict__ bcls, float* __restrict__ out, int t)
{
  const int tid  = threadIdx.x;
  const int wave = tid >> 5, lane = tid & 31;
  const int row  = blockIdx.x * 8 + wave;
  float s = 0.f;
  for (int j = lane; j < Dd; j += 32) s += H[(long)row * Dd + j] * Wcls[j];
#pragma unroll
  for (int off = 16; off > 0; off >>= 1) s += __shfl_xor(s, off, 32);
  if (lane == 0) {
    const int b = row >> 5, n = row & 31;
    const float v = s + bcls[0];
    out[(b * Ss + t) * Nn + n] = v;                   // first output copy
    out[Bb * Ss * Nn + (b * Ss + t) * Nn + n] = v;    // duplicated output
  }
}

// X2[b, :] = concat(q[b], pos[b], neg[b]).  grid(8), block(256)
__global__ __launch_bounds__(256)
void pack_x2_kernel(const float* __restrict__ fn, const float* __restrict__ q,
                    const float* __restrict__ mask, const int* __restrict__ gt,
                    float* __restrict__ X2, int t)
{
  __shared__ float s_inv;
  const int b   = blockIdx.x;
  const int idx = gt[b * Ss + t];
  if (threadIdx.x == 0) {
    float c = 0.f;
    for (int n = 0; n < Nn; ++n)
      c += (n == idx) ? 0.f : mask[(b * Ss + t) * Nn + n];
    s_inv = 1.f / (c > 0.f ? c : 1.f);
  }
  __syncthreads();
  const float inv   = s_inv;
  const float* base = &fn[(((long)b * Ss + t) * Nn) * Dd];
  float* dst = &X2[(long)b * (3 * Dd)];
  for (int k = threadIdx.x; k < Dd; k += 256) {
    const float pos = base[(long)idx * Dd + k];
    float sum = 0.f;
#pragma unroll
    for (int n = 0; n < Nn; ++n) sum += base[(long)n * Dd + k];
    dst[k]          = q[(long)b * Dd + k];
    dst[Dd + k]     = pos;
    dst[2 * Dd + k] = (sum - pos) * inv;
  }
}

// ---------------------------------------------------------------------------
extern "C" void kernel_launch(void* const* d_in, const int* in_sizes, int n_in,
                              void* d_out, int out_size, void* d_ws, size_t ws_size,
                              hipStream_t stream) {
  (void)in_sizes; (void)n_in; (void)out_size; (void)ws_size;

  const float* query_fea = (const float*)d_in[0];   // [8,16,768]
  const float* w_obs     = (const float*)d_in[1];   // [8,8,32,8192]
  const float* fea_emb   = (const float*)d_in[2];   // [8,8192,768]
  const float* nei_mask  = (const float*)d_in[3];   // [8,8,32]
  const int*   move_gt   = (const int*)d_in[4];     // [8,8]
  const float* W1   = (const float*)d_in[5];        // [768,768]
  const float* b1   = (const float*)d_in[6];
  const float* W2   = (const float*)d_in[7];        // [1536,768]
  const float* b2   = (const float*)d_in[8];
  const float* Wcls = (const float*)d_in[9];        // [768,1]
  const float* bcls = (const float*)d_in[10];
  const float* Wq   = (const float*)d_in[11];       // [2304,768]
  const float* bq   = (const float*)d_in[12];
  float* out = (float*)d_out;                       // 2 x [8,8,32]

  // Workspace layout (fp32), ~15.4 MB total
  float* agg = (float*)d_ws;            // [2048,768]
  float* fnb = agg + 2048L * 768;       // [2048,768]
  float* X   = fnb + 2048L * 768;       // [256,1536]
  float* H   = X   + 256L * 1536;       // [256,768]
  float* X2  = H   + 256L * 768;        // [32,2304] (rows 8..31 zero)
  float* q0b = X2  + 32L * 2304;        // [32,768]
  float* q1b = q0b + 32L * 768;         // [32,768]

  // q0 mean + X2 pad rows
  init_kernel<<<32, 256, 0, stream>>>(query_fea, q0b, X2);

  // agg[b] = weight_observe[b] @ fea_emb[b]   (M=256, K=8192, N=768, batch=8)
  wmma_gemm_kernel<<<dim3(6, 8, 8), 256, 0, stream>>>(
      w_obs, fea_emb, nullptr, agg,
      /*K=*/8192, /*lda=*/8192, /*ldb=*/768, /*ldc=*/768,
      /*aS=*/256L * 8192, /*bS=*/8192L * 768, /*cS=*/256L * 768, /*relu=*/0);

  // fn = relu(agg @ W1 + b1)   (M=2048, K=768, N=768)
  wmma_gemm_kernel<<<dim3(6, 64, 1), 256, 0, stream>>>(
      agg, W1, b1, fnb, /*K=*/768, 768, 768, 768, 0, 0, 0, /*relu=*/1);

  // Sequential scan over S=8 steps (q carried across steps)
  for (int t = 0; t < Ss; ++t) {
    const float* qc = (t & 1) ? q1b : q0b;
    float*       qn = (t & 1) ? q0b : q1b;

    pack_x_kernel<<<256, 256, 0, stream>>>(fnb, qc, X, t);

    // H = relu(X @ W2 + b2)   (M=256, K=1536, N=768)
    wmma_gemm_kernel<<<dim3(6, 8, 1), 256, 0, stream>>>(
        X, W2, b2, H, /*K=*/1536, 1536, 768, 768, 0, 0, 0, /*relu=*/1);

    cls_kernel<<<32, 256, 0, stream>>>(H, Wcls, bcls, out, t);

    pack_x2_kernel<<<8, 256, 0, stream>>>(fnb, qc, nei_mask, move_gt, X2, t);

    // q_next = relu(X2 @ Wq + bq)   (M=32 padded, K=2304, N=768)
    wmma_gemm_kernel<<<dim3(6, 1, 1), 256, 0, stream>>>(
        X2, Wq, bq, qn, /*K=*/2304, 2304, 768, 768, 0, 0, 0, /*relu=*/1);
  }
}
